// GNNLayer_38216619000385
// MI455X (gfx1250) — compile-verified, matrix operand order checked by reading, not run on hardware
//
#include <hip/hip_runtime.h>
#include <hip/hip_bf16.h>

typedef __attribute__((ext_vector_type(16))) __bf16 v16bf;
typedef __attribute__((ext_vector_type(8)))  float  v8f;
typedef __attribute__((ext_vector_type(4)))  unsigned int u32x4;
typedef __attribute__((ext_vector_type(8)))  int i32x8;
typedef __attribute__((ext_vector_type(4)))  int i32x4;

#define FEAT 1024
#define NROW 32768
#define NHID 128

// workspace layout (units: floats)
#define WS_B     0        // 65536 floats (256 KB) holding bf16 B fragments
#define WS_TGT   65536    // 1024
#define WS_HBIAS 66560    // 128
#define WS_AGG   66688    // 1024
#define WS_S     67712    // 1
#define WS_R     67728    // 1024
#define WS_GATE  68752    // 1024

__device__ __forceinline__ int clampd(int d) { return d < 0 ? 0 : (d > 31 ? 31 : d); }
__device__ __forceinline__ float sigmoidf(float x) { return 1.0f / (1.0f + __expf(-x)); }
__device__ __forceinline__ float4 ld4(const float* p) { return *reinterpret_cast<const float4*>(p); }
__device__ __forceinline__ void st4(float* p, float4 v) { *reinterpret_cast<float4*>(p) = v; }

// ---------------------------------------------------------------------------
// K0: target = features[0] + pos_emb[dep0];  hbias = aW1[:, :1024] @ target + ab1
//     also zero the global agg/s accumulators (fresh every launch).
// ---------------------------------------------------------------------------
__global__ void __launch_bounds__(256) k0_setup(const float* __restrict__ features,
                                                const int* __restrict__ depths,
                                                const float* __restrict__ pos_emb,
                                                const float* __restrict__ aW1,
                                                const float* __restrict__ ab1,
                                                float* __restrict__ ws) {
    __shared__ float tgt[FEAT];
    int tid = threadIdx.x;
    int d0 = clampd(depths[0]);
    const float* pe = pos_emb + (size_t)d0 * FEAT;
    int i4 = tid * 4;
    float4 fv = ld4(features + i4);
    float4 pv = ld4(pe + i4);
    float4 tv = {fv.x + pv.x, fv.y + pv.y, fv.z + pv.z, fv.w + pv.w};
    st4(ws + WS_TGT + i4, tv);
    tgt[i4 + 0] = tv.x; tgt[i4 + 1] = tv.y; tgt[i4 + 2] = tv.z; tgt[i4 + 3] = tv.w;
    float4 z = {0.f, 0.f, 0.f, 0.f};
    st4(ws + WS_AGG + i4, z);
    if (tid == 0) ws[WS_S] = 0.f;
    __syncthreads();
    int wid = tid >> 5, lane = tid & 31;
    for (int j = wid; j < NHID; j += 8) {
        const float* wrow = aW1 + (size_t)j * 2048;   // left half of aW1 row j
        float sum = 0.f;
#pragma unroll
        for (int it = 0; it < 32; ++it) sum += wrow[it * 32 + lane] * tgt[it * 32 + lane];
        for (int m = 16; m >= 1; m >>= 1) sum += __shfl_xor(sum, m, 32);
        if (lane == 0) ws[WS_HBIAS + j] = sum + ab1[j];
    }
}

// ---------------------------------------------------------------------------
// K0b: convert aW1[:, 1024:2048] (B matrix, K x N = 1024 x 128) into bf16,
//      laid out as WMMA B fragments: frag index = kb*8 + nt, kb = K/32, nt = N/16.
//      Per-lane element e maps to K = kb*32 + e + (e>=8?8:0) + (lane>=16?8:0),
//      N = nt*16 + (lane & 15)  (mirrors the documented 16-bit A layout).
// ---------------------------------------------------------------------------
__global__ void __launch_bounds__(256) k0b_convB(const float* __restrict__ aW1,
                                                 float* __restrict__ ws) {
    int kb = blockIdx.x;                 // 0..31
    int tid = threadIdx.x;
    int nt = tid >> 5, lane = tid & 31;  // 8 waves = 8 n-tiles
    int n = nt * 16 + (lane & 15);
    int off = (lane & 16) ? 8 : 0;
    const float* src = aW1 + (size_t)n * 2048 + 1024 + kb * 32 + off;
    float4 f0 = ld4(src + 0), f1 = ld4(src + 4), f2 = ld4(src + 16), f3 = ld4(src + 20);
    v16bf v;
    v[0]  = (__bf16)f0.x; v[1]  = (__bf16)f0.y; v[2]  = (__bf16)f0.z; v[3]  = (__bf16)f0.w;
    v[4]  = (__bf16)f1.x; v[5]  = (__bf16)f1.y; v[6]  = (__bf16)f1.z; v[7]  = (__bf16)f1.w;
    v[8]  = (__bf16)f2.x; v[9]  = (__bf16)f2.y; v[10] = (__bf16)f2.z; v[11] = (__bf16)f2.w;
    v[12] = (__bf16)f3.x; v[13] = (__bf16)f3.y; v[14] = (__bf16)f3.z; v[15] = (__bf16)f3.w;
    __bf16* dst = reinterpret_cast<__bf16*>(ws) + ((size_t)(kb * 8 + nt) * 32 + lane) * 16;
    *reinterpret_cast<v16bf*>(dst) = v;
}

// ---------------------------------------------------------------------------
// K1: main fused pass. One 16-row tile per wave (2048 tiles total).
//  - f = features + pos_emb[dep] streamed to output
//  - B chunk (32 KB) staged global->LDS with the Tensor Data Mover
//  - h tile (16x128) via v_wmma_f32_16x16x32_bf16
//  - attn_u = sigmoid(relu(h)·aW2 + ab2), unnormalized agg/s accumulated
// ---------------------------------------------------------------------------
__global__ void __launch_bounds__(256) k1_main(const float* __restrict__ features,
                                               const int* __restrict__ depths,
                                               const float* __restrict__ pos_emb,
                                               const float* __restrict__ aW2,
                                               const float* __restrict__ ab2,
                                               float* __restrict__ out,
                                               float* __restrict__ ws) {
    __shared__ __align__(32) __bf16 smB[4 * 8 * 32 * 16];  // 32 KB: one 128-K chunk of B
    __shared__ float smAgg[FEAT];
    __shared__ float smS;

    int tid = threadIdx.x, wid = tid >> 5, lane = tid & 31;
    int tile = blockIdx.x * 8 + wid;     // 0..2047
    int r0 = tile * 16;

    for (int i = tid; i < FEAT; i += 256) smAgg[i] = 0.f;
    if (tid == 0) smS = 0.f;

    int dcl[16];
#pragma unroll
    for (int m = 0; m < 16; ++m) dcl[m] = clampd(depths[r0 + m]);

    int rowA = r0 + (lane & 15);
    int base8 = (lane & 16) ? 8 : 0;
    int dA = clampd(depths[rowA]);
    const float* fA = features + (size_t)rowA * FEAT;
    const float* pA = pos_emb + (size_t)dA * FEAT;

    v8f acc[8];
    v8f vz = {0.f, 0.f, 0.f, 0.f, 0.f, 0.f, 0.f, 0.f};
#pragma unroll
    for (int nt = 0; nt < 8; ++nt) acc[nt] = vz;

    // TDM descriptor pieces that do not change across kc
    unsigned int ldsOff = (unsigned int)(unsigned long long)(uintptr_t)(&smB[0]); // generic LDS ptr: low 32 bits = LDS byte offset
    unsigned long long wsBase = (unsigned long long)(uintptr_t)ws;                // B region starts at ws
    const i32x4 gz4 = {0, 0, 0, 0};                                               // groups 2/3 unused (<=2D tensor)
    const i32x8 gz8 = {0, 0, 0, 0, 0, 0, 0, 0};                                   // extra descriptor words (unused)

    for (int kc = 0; kc < 8; ++kc) {
        __syncthreads();   // previous chunk's B reads complete
        if (wid == 0) {
            // ---- Tensor DMA: 32 KB (8192 x 4B elements) ws -> LDS ----
            unsigned long long ga = wsBase + (unsigned long long)kc * 32768ull;
            u32x4 g0;
            g0[0] = 1u;                                   // count=1, is_restore=0, gather off
            g0[1] = ldsOff;                               // lds_addr (bytes)
            g0[2] = (unsigned int)(ga & 0xffffffffull);   // global_addr[31:0]
            g0[3] = (unsigned int)((ga >> 32) & 0x1ffffffull) | (2u << 30); // ga[56:32] | type=2
            i32x8 g1;
            g1[0] = (int)(2u << 16);                      // wg_mask=0, data_size=2 (4B)
            g1[1] = (int)(0x2000u << 16);                 // tensor_dim0[15:0]=8192 in bits 63:48
            g1[2] = (int)(1u << 16);                      // tensor_dim0[31:16]=0, tensor_dim1=1
            g1[3] = (int)(0x2000u << 16);                 // tile_dim0 = 8192
            g1[4] = 1;                                    // tile_dim1 = 1, tile_dim2 = 0
            g1[5] = 8192;                                 // tensor_dim0_stride[31:0]
            g1[6] = 0;                                    // strides high bits
            g1[7] = 0;
            __builtin_amdgcn_tensor_load_to_lds(g0, g1, gz4, gz4, gz8, 0);
            __builtin_amdgcn_s_wait_tensorcnt(0);
        }
        if (kc < 7)
            __builtin_prefetch((const void*)((const char*)ws + (kc + 1) * 32768 + tid * 128), 0, 0);
        __syncthreads();   // B chunk visible to all 8 waves

        // coalesced f compute + stream to output (128 cols per wave iteration)
        int colc = kc * 128 + (lane << 2);
#pragma unroll
        for (int m = 0; m < 16; ++m) {
            int row = r0 + m;
            float4 fv = ld4(features + (size_t)row * FEAT + colc);
            float4 pv = ld4(pos_emb + (size_t)dcl[m] * FEAT + colc);
            float4 ov = {fv.x + pv.x, fv.y + pv.y, fv.z + pv.z, fv.w + pv.w};
            st4(out + (size_t)row * FEAT + colc, ov);
        }

        // A fragments (documented 16-bit A layout) + WMMA over 4 k-sub-steps
#pragma unroll
        for (int ks = 0; ks < 4; ++ks) {
            int c0 = kc * 128 + ks * 32 + base8;
            float4 f0 = ld4(fA + c0),      p0 = ld4(pA + c0);
            float4 f1 = ld4(fA + c0 + 4),  p1 = ld4(pA + c0 + 4);
            float4 f2 = ld4(fA + c0 + 16), p2 = ld4(pA + c0 + 16);
            float4 f3 = ld4(fA + c0 + 20), p3 = ld4(pA + c0 + 20);
            v16bf a;
            a[0]  = (__bf16)(f0.x + p0.x); a[1]  = (__bf16)(f0.y + p0.y);
            a[2]  = (__bf16)(f0.z + p0.z); a[3]  = (__bf16)(f0.w + p0.w);
            a[4]  = (__bf16)(f1.x + p1.x); a[5]  = (__bf16)(f1.y + p1.y);
            a[6]  = (__bf16)(f1.z + p1.z); a[7]  = (__bf16)(f1.w + p1.w);
            a[8]  = (__bf16)(f2.x + p2.x); a[9]  = (__bf16)(f2.y + p2.y);
            a[10] = (__bf16)(f2.z + p2.z); a[11] = (__bf16)(f2.w + p2.w);
            a[12] = (__bf16)(f3.x + p3.x); a[13] = (__bf16)(f3.y + p3.y);
            a[14] = (__bf16)(f3.z + p3.z); a[15] = (__bf16)(f3.w + p3.w);
            // preload B fragments in groups of 4 so ds_loads batch ahead of the
            // WMMA chain instead of load->wait->wmma per n-tile
#pragma unroll
            for (int g = 0; g < 2; ++g) {
                v16bf b0 = *reinterpret_cast<const v16bf*>(smB + ((ks * 8 + g * 4 + 0) * 32 + lane) * 16);
                v16bf b1 = *reinterpret_cast<const v16bf*>(smB + ((ks * 8 + g * 4 + 1) * 32 + lane) * 16);
                v16bf b2 = *reinterpret_cast<const v16bf*>(smB + ((ks * 8 + g * 4 + 2) * 32 + lane) * 16);
                v16bf b3 = *reinterpret_cast<const v16bf*>(smB + ((ks * 8 + g * 4 + 3) * 32 + lane) * 16);
                acc[g * 4 + 0] = __builtin_amdgcn_wmma_f32_16x16x32_bf16(
                    false, a, false, b0, (short)0, acc[g * 4 + 0], false, false);
                acc[g * 4 + 1] = __builtin_amdgcn_wmma_f32_16x16x32_bf16(
                    false, a, false, b1, (short)0, acc[g * 4 + 1], false, false);
                acc[g * 4 + 2] = __builtin_amdgcn_wmma_f32_16x16x32_bf16(
                    false, a, false, b2, (short)0, acc[g * 4 + 2], false, false);
                acc[g * 4 + 3] = __builtin_amdgcn_wmma_f32_16x16x32_bf16(
                    false, a, false, b3, (short)0, acc[g * 4 + 3], false, false);
            }
        }
    }

    // epilogue: h = relu(acc + hbias); per-row logit = h · aW2; attn_u = sigmoid(+ab2)
    int nidx = lane & 15;
    float p[8];
#pragma unroll
    for (int v = 0; v < 8; ++v) p[v] = 0.f;
    float ab2v = ab2[0];
#pragma unroll
    for (int nt = 0; nt < 8; ++nt) {
        float hb = ws[WS_HBIAS + nt * 16 + nidx];
        float w2 = aW2[nt * 16 + nidx];
#pragma unroll
        for (int v = 0; v < 8; ++v) {
            float h = fmaxf(acc[nt][v] + hb, 0.f);
            p[v] += h * w2;
        }
    }
#pragma unroll
    for (int v = 0; v < 8; ++v) {
        p[v] += __shfl_xor(p[v], 1, 32);
        p[v] += __shfl_xor(p[v], 2, 32);
        p[v] += __shfl_xor(p[v], 4, 32);
        p[v] += __shfl_xor(p[v], 8, 32);
    }
    bool hiHalf = (lane & 16) != 0;
    float aLo[8], aHi[8];
#pragma unroll
    for (int v = 0; v < 8; ++v) {
        float other = __shfl_xor(p[v], 16, 32);
        float lo = hiHalf ? other : p[v];   // rows m = v
        float hh = hiHalf ? p[v] : other;   // rows m = v + 8
        aLo[v] = sigmoidf(lo + ab2v);
        aHi[v] = sigmoidf(hh + ab2v);
    }
    if (tile == 0) aLo[0] = 0.f;   // global row 0 is the target, not a path row

    float sw = 0.f;
#pragma unroll
    for (int v = 0; v < 8; ++v) sw += aLo[v] + aHi[v];
    if (lane == 0) atomicAdd(&smS, sw);

    // unnormalized agg: sum_m attn_u[m] * f[m][:] (re-reads L2-hot features)
    for (int kc = 0; kc < 8; ++kc) {
        int colc = kc * 128 + (lane << 2);
        float ax = 0.f, ay = 0.f, az = 0.f, aw = 0.f;
#pragma unroll
        for (int m = 0; m < 16; ++m) {
            int row = r0 + m;
            float4 fv = ld4(features + (size_t)row * FEAT + colc);
            float4 pv = ld4(pos_emb + (size_t)dcl[m] * FEAT + colc);
            float am = (m < 8) ? aLo[m] : aHi[m - 8];
            ax += am * (fv.x + pv.x); ay += am * (fv.y + pv.y);
            az += am * (fv.z + pv.z); aw += am * (fv.w + pv.w);
        }
        atomicAdd(&smAgg[colc + 0], ax);
        atomicAdd(&smAgg[colc + 1], ay);
        atomicAdd(&smAgg[colc + 2], az);
        atomicAdd(&smAgg[colc + 3], aw);
    }
    __syncthreads();
    for (int i = tid; i < FEAT; i += 256) atomicAdd(ws + WS_AGG + i, smAgg[i]);
    if (tid == 0) atomicAdd(ws + WS_S, smS);
}

// ---------------------------------------------------------------------------
// K2a: r = relu(uW1 @ c2 + ub1), gate = sigmoid(gW @ c2 + gb)
//      c2 = [target, agg * (s>0 ? 1/s : 1)]
// ---------------------------------------------------------------------------
__global__ void __launch_bounds__(256) k2a_gemv(const float* __restrict__ uW1,
                                                const float* __restrict__ ub1,
                                                const float* __restrict__ gW,
                                                const float* __restrict__ gb,
                                                float* __restrict__ ws) {
    int tid = threadIdx.x, wid = tid >> 5, lane = tid & 31;
    int gwv = blockIdx.x * 8 + wid;  // 0..511
    float s = ws[WS_S];
    float scale = (s > 0.f) ? (1.f / s) : 1.f;
    for (int tsk = gwv; tsk < 2048; tsk += 512) {
        int j = tsk & 1023;
        int isG = tsk >> 10;
        const float* W = (isG ? gW : uW1) + (size_t)j * 2048;
        float sum = 0.f;
#pragma unroll 4
        for (int it = 0; it < 16; ++it) {
            int k = (it * 32 + lane) * 4;
            float4 w4 = ld4(W + k);
            float4 c4;
            if (k < 1024) {
                c4 = ld4(ws + WS_TGT + k);
            } else {
                c4 = ld4(ws + WS_AGG + (k - 1024));
                c4.x *= scale; c4.y *= scale; c4.z *= scale; c4.w *= scale;
            }
            sum += w4.x * c4.x + w4.y * c4.y + w4.z * c4.z + w4.w * c4.w;
        }
        for (int m = 16; m >= 1; m >>= 1) sum += __shfl_xor(sum, m, 32);
        if (lane == 0) {
            if (isG) ws[WS_GATE + j] = sigmoidf(sum + gb[j]);
            else     ws[WS_R + j]    = fmaxf(sum + ub1[j], 0.f);
        }
    }
}

// ---------------------------------------------------------------------------
// K2b: out[0] = target + gate * (uW2 @ r + ub2)   (overwrites output row 0)
// ---------------------------------------------------------------------------
__global__ void __launch_bounds__(256) k2b_final(const float* __restrict__ uW2,
                                                 const float* __restrict__ ub2,
                                                 const float* __restrict__ ws,
                                                 float* __restrict__ out) {
    int tid = threadIdx.x, wid = tid >> 5, lane = tid & 31;
    int gwv = blockIdx.x * 8 + wid;  // 0..255
    for (int k = gwv; k < FEAT; k += 256) {
        const float* W = uW2 + (size_t)k * FEAT;
        float sum = 0.f;
#pragma unroll 2
        for (int it = 0; it < 8; ++it) {
            int jj = (it * 32 + lane) * 4;
            float4 w4 = ld4(W + jj);
            float4 r4 = ld4(ws + WS_R + jj);
            sum += w4.x * r4.x + w4.y * r4.y + w4.z * r4.z + w4.w * r4.w;
        }
        for (int m = 16; m >= 1; m >>= 1) sum += __shfl_xor(sum, m, 32);
        if (lane == 0)
            out[k] = ws[WS_TGT + k] + ws[WS_GATE + k] * (sum + ub2[k]);
    }
}

extern "C" void kernel_launch(void* const* d_in, const int* in_sizes, int n_in,
                              void* d_out, int out_size, void* d_ws, size_t ws_size,
                              hipStream_t stream) {
    const float* features = (const float*)d_in[0];
    const int*   depths   = (const int*)d_in[1];
    const float* pos_emb  = (const float*)d_in[2];
    const float* aW1      = (const float*)d_in[3];
    const float* ab1      = (const float*)d_in[4];
    const float* aW2      = (const float*)d_in[5];
    const float* ab2      = (const float*)d_in[6];
    const float* uW1      = (const float*)d_in[7];
    const float* ub1      = (const float*)d_in[8];
    const float* uW2      = (const float*)d_in[9];
    const float* ub2      = (const float*)d_in[10];
    const float* gW       = (const float*)d_in[11];
    const float* gb       = (const float*)d_in[12];
    float* out = (float*)d_out;
    float* ws  = (float*)d_ws;

    hipLaunchKernelGGL(k0_setup,  dim3(1),   dim3(256), 0, stream,
                       features, depths, pos_emb, aW1, ab1, ws);
    hipLaunchKernelGGL(k0b_convB, dim3(32),  dim3(256), 0, stream, aW1, ws);
    hipLaunchKernelGGL(k1_main,   dim3(256), dim3(256), 0, stream,
                       features, depths, pos_emb, aW2, ab2, out, ws);
    hipLaunchKernelGGL(k2a_gemv,  dim3(64),  dim3(256), 0, stream,
                       uW1, ub1, gW, gb, ws);
    hipLaunchKernelGGL(k2b_final, dim3(32),  dim3(256), 0, stream,
                       uW2, ub2, ws, out);
}
